// MultiHeadAttention_2095944041211
// MI455X (gfx1250) — matrix operand order repaired and checked by reading, not compiled, not run on hardware
//
#include <hip/hip_runtime.h>

// ---------------------------------------------------------------------------
// MHA forward for gfx1250 (MI455X): bf16 WMMA GEMMs (f32 accumulate),
// TDM-staged K-tiles in LDS for the scores GEMM.
// d_out = [ out : 4096*1024 f32 ][ attn : 32*2048*2048 f32 ]
// ---------------------------------------------------------------------------

#define D_MODEL 1024
#define SEQ     2048
#define NBATCH  2
#define NHEADS  16
#define DK      64
#define MTOT    (NBATCH * SEQ)          // 4096 rows of Q/K/V/ctx

typedef __attribute__((ext_vector_type(16))) __bf16 bf16x16;
typedef __attribute__((ext_vector_type(8)))  __bf16 bf16x8;
typedef __attribute__((ext_vector_type(4)))  __bf16 bf16x4;
typedef __attribute__((ext_vector_type(8)))  float  f32x8;
typedef __attribute__((ext_vector_type(4)))  float  f32x4;
typedef __attribute__((ext_vector_type(4)))  unsigned int u32x4;
typedef __attribute__((ext_vector_type(8)))  int    i32x8;
typedef __attribute__((ext_vector_type(4)))  int    i32x4;

#define WMMA_BF16(acc, a, b) \
    __builtin_amdgcn_wmma_f32_16x16x32_bf16(false, (a), false, (b), (short)0, (acc), false, false)

// ---- WMMA fragment loaders (contiguous k order, same for A and B, wave32) ----------

// A fragment 16x32 bf16: lane L: M = m0 + L%16 ;
//   elems 0..15 -> K = k0 + 16*(L/16) + i
static __device__ inline bf16x16 load_A_bf16(const __bf16* __restrict__ A, int lda,
                                             int m0, int k0, int lane) {
    const int m  = m0 + (lane & 15);
    const int kb = k0 + ((lane >> 4) << 4);
    const __bf16* p = A + (size_t)m * lda + kb;
    bf16x8 lo = *(const bf16x8*)p;
    bf16x8 hi = *(const bf16x8*)(p + 8);
    bf16x16 r;
#pragma unroll
    for (int i = 0; i < 8; ++i) { r[i] = lo[i]; r[8 + i] = hi[i]; }
    return r;
}

// ---------------------------------------------------------------------------
// Kernel 0: f32 -> bf16 bulk convert (x and the four weight matrices).
// ---------------------------------------------------------------------------
__global__ __launch_bounds__(256) void k_cvt_bf16(const float* __restrict__ src,
                                                  __bf16* __restrict__ dst, int n4)
{
    const int i = blockIdx.x * 256 + threadIdx.x;
    if (i < n4) {
        f32x4 v = *(const f32x4*)(src + 4 * (size_t)i);
        bf16x4 o;
        o[0] = (__bf16)v[0]; o[1] = (__bf16)v[1];
        o[2] = (__bf16)v[2]; o[3] = (__bf16)v[3];
        *(bf16x4*)(dst + 4 * (size_t)i) = o;
    }
}

// ---------------------------------------------------------------------------
// Kernel 1: QKV projection.  y = x @ W.T + b  (all-bf16 operands)
// blockIdx.z: 0=Q 1=K 2=V.  Q,K row-major bf16; V stored transposed (Vt[d][m]).
// Wave: 16x64 tile (4 accumulators); block of 4 waves covers 256 N.
// ---------------------------------------------------------------------------
__global__ __launch_bounds__(128) void k_qkv_proj(
    const __bf16* __restrict__ xb,
    const __bf16* __restrict__ wqb, const float* __restrict__ bq,
    const __bf16* __restrict__ wkb, const float* __restrict__ bk,
    const __bf16* __restrict__ wvb, const float* __restrict__ bv,
    __bf16* __restrict__ Qb, __bf16* __restrict__ Kb, __bf16* __restrict__ Vt)
{
    const int lane  = threadIdx.x & 31;
    const int wave  = threadIdx.x >> 5;
    const int m0    = blockIdx.x * 16;
    const int n0    = blockIdx.y * 256 + wave * 64;
    const int which = blockIdx.z;

    const __bf16* __restrict__ W    = (which == 0) ? wqb : (which == 1) ? wkb : wvb;
    const float*  __restrict__ bias = (which == 0) ? bq  : (which == 1) ? bk  : bv;

    const int nlo  = lane & 15;
    const int half = lane >> 4;

    f32x8 acc0 = {}, acc1 = {}, acc2 = {}, acc3 = {};
    const __bf16* wrow = W + (size_t)(n0 + nlo) * D_MODEL + (half << 4);
    for (int k0 = 0; k0 < D_MODEL; k0 += 32) {
        bf16x16 a  = load_A_bf16(xb, D_MODEL, m0, k0, lane);
        bf16x16 b0 = *(const bf16x16*)(wrow + k0);
        bf16x16 b1 = *(const bf16x16*)(wrow + k0 + 16 * D_MODEL);
        bf16x16 b2 = *(const bf16x16*)(wrow + k0 + 32 * D_MODEL);
        bf16x16 b3 = *(const bf16x16*)(wrow + k0 + 48 * D_MODEL);
        acc0 = WMMA_BF16(acc0, a, b0);
        acc1 = WMMA_BF16(acc1, a, b1);
        acc2 = WMMA_BF16(acc2, a, b2);
        acc3 = WMMA_BF16(acc3, a, b3);
    }

    const int mb = m0 + (half << 3);
    f32x8 accs[4] = {acc0, acc1, acc2, acc3};
#pragma unroll
    for (int t = 0; t < 4; ++t) {
        const int   n  = n0 + 16 * t + nlo;
        const float bb = bias[n];
#pragma unroll
        for (int r = 0; r < 8; ++r) {
            const float v = accs[t][r] + bb;
            if (which == 2)      Vt[(size_t)n * MTOT + (mb + r)]    = (__bf16)v;
            else if (which == 1) Kb[(size_t)(mb + r) * D_MODEL + n] = (__bf16)v;
            else                 Qb[(size_t)(mb + r) * D_MODEL + n] = (__bf16)v;
        }
    }
}

// ---------------------------------------------------------------------------
// Kernel 2: scores = (Q Kᵀ)/8 per (b,h), raw into attn region of d_out.
// One TDM tensor_load_to_lds stages the 256x64 bf16 K-tile for the whole
// block (4 waves x 64 kk each), with 16B row padding (144B LDS row stride)
// for bank-conflict-free fragment reads.
// ---------------------------------------------------------------------------
#define KT_ROW 72            // 64 bf16 + 8 bf16 (16B) pad = 144B row stride

__global__ __launch_bounds__(128) void k_scores(
    const __bf16* __restrict__ Qb, const __bf16* __restrict__ Kb,
    float* __restrict__ attn)
{
    __shared__ __bf16 ktile[256 * KT_ROW];      // 36 KB

    const int lane  = threadIdx.x & 31;
    const int wave  = threadIdx.x >> 5;
    const int bh    = blockIdx.z;
    const int b     = bh >> 4;
    const int h     = bh & 15;
    const int q0    = blockIdx.x * 16;
    const int kkblk = blockIdx.y * 256;

    const __bf16* Qp = Qb + (size_t)b * SEQ * D_MODEL;
    const __bf16* Kp = Kb + (size_t)b * SEQ * D_MODEL;

    if (wave == 0) {
        // --- Tensor DMA descriptor (CDNA5 D#, 2D tile 64 x 256 rows) -------
        const __bf16* gsrc = Kp + (size_t)kkblk * D_MODEL + h * DK;
        unsigned long long ga = (unsigned long long)(uintptr_t)gsrc;
        unsigned lds = (unsigned)(uintptr_t)(&ktile[0]);   // low 32 = LDS offset
        u32x4 g0;
        g0[0] = 1u;                                        // count=1 (valid), user mode
        g0[1] = lds;                                       // lds_addr
        g0[2] = (unsigned)(ga & 0xffffffffull);            // global_addr[31:0]
        g0[3] = (unsigned)((ga >> 32) & 0x1ffffffull)      // global_addr[56:32]
              | (2u << 30);                                // type = 2 ("image")
        i32x8 g1;
        g1[0] = (1 << 16)                                  // data_size = 2 bytes
              | (1 << 20)                                  // pad_enable
              | (4 << 22)                                  // pad_interval = 32 DW (128B)
              | (3 << 25);                                 // pad_amount   = 4 DW (16B)
        g1[1] = (D_MODEL & 0xffff) << 16;                  // tensor_dim0 = 1024
        g1[2] = (SEQ & 0xffff) << 16;                      // tensor_dim1 = 2048
        g1[3] = (DK << 16);                                // tile_dim0 = 64
        g1[4] = 256;                                       // tile_dim1 = 256, tile_dim2 = 0
        g1[5] = D_MODEL;                                   // tensor_dim0_stride = 1024
        g1[6] = 0;
        g1[7] = 0;
        i32x4 gz4 = {0, 0, 0, 0};
        i32x8 gz8 = {0, 0, 0, 0, 0, 0, 0, 0};
        __builtin_amdgcn_tensor_load_to_lds(g0, g1, gz4, gz4, gz8, 0);
        __builtin_amdgcn_s_wait_tensorcnt((short)0);
    }
    __syncthreads();

    const int nlo  = lane & 15;
    const int half = lane >> 4;
    const int rowb = wave * 64;                 // this wave's kk sub-block in LDS

    f32x8 acc0 = {}, acc1 = {}, acc2 = {}, acc3 = {};
#pragma unroll
    for (int kc = 0; kc < DK; kc += 32) {
        bf16x16 a  = load_A_bf16(Qp, D_MODEL, q0, h * DK + kc, lane);
        const __bf16* l = ktile + (size_t)(rowb + nlo) * KT_ROW + kc + (half << 4);
        bf16x16 b0 = *(const bf16x16*)(l);
        bf16x16 b1 = *(const bf16x16*)(l + 16 * KT_ROW);
        bf16x16 b2 = *(const bf16x16*)(l + 32 * KT_ROW);
        bf16x16 b3 = *(const bf16x16*)(l + 48 * KT_ROW);
        acc0 = WMMA_BF16(acc0, a, b0);
        acc1 = WMMA_BF16(acc1, a, b1);
        acc2 = WMMA_BF16(acc2, a, b2);
        acc3 = WMMA_BF16(acc3, a, b3);
    }

    float* out = attn + (size_t)bh * SEQ * SEQ;
    const int mb = q0 + (half << 3);
    const float scale = 0.125f;                 // 1/sqrt(64)
    const int n0 = kkblk + wave * 64;
    f32x8 accs[4] = {acc0, acc1, acc2, acc3};
#pragma unroll
    for (int t = 0; t < 4; ++t) {
        const int n = n0 + 16 * t + nlo;
#pragma unroll
        for (int r = 0; r < 8; ++r)
            out[(size_t)(mb + r) * SEQ + n] = accs[t][r] * scale;
    }
}

// ---------------------------------------------------------------------------
// Kernel 3: row softmax in place (256 threads/row, row cached in registers).
// ---------------------------------------------------------------------------
__global__ __launch_bounds__(256) void k_softmax(float* __restrict__ attn)
{
    __shared__ float red[256];
    const int t = threadIdx.x;
    float* row = attn + (size_t)blockIdx.x * SEQ;

    float v[8];
#pragma unroll
    for (int i = 0; i < 8; ++i) v[i] = row[t + 256 * i];

    float m = v[0];
#pragma unroll
    for (int i = 1; i < 8; ++i) m = fmaxf(m, v[i]);
    red[t] = m;
    __syncthreads();
    for (int off = 128; off > 0; off >>= 1) {
        if (t < off) red[t] = fmaxf(red[t], red[t + off]);
        __syncthreads();
    }
    m = red[0];
    __syncthreads();

    float s = 0.f;
#pragma unroll
    for (int i = 0; i < 8; ++i) { v[i] = __expf(v[i] - m); s += v[i]; }
    red[t] = s;
    __syncthreads();
    for (int off = 128; off > 0; off >>= 1) {
        if (t < off) red[t] += red[t + off];
        __syncthreads();
    }
    const float inv = 1.0f / red[0];
#pragma unroll
    for (int i = 0; i < 8; ++i) row[t + 256 * i] = v[i] * inv;
}

// ---------------------------------------------------------------------------
// Kernel 4: ctx = attn @ V per (b,h).  Raw f32 probability loads are batched
// into float4 temporaries before conversion so the compiler can cluster the
// global loads; B fragments are contiguous 32B bf16 loads from transposed V.
// ---------------------------------------------------------------------------
__global__ __launch_bounds__(128) void k_pv(
    const float* __restrict__ attn, const __bf16* __restrict__ Vt,
    __bf16* __restrict__ Cx)
{
    const int lane = threadIdx.x & 31;
    const int wave = threadIdx.x >> 5;
    const int bh   = blockIdx.y;
    const int b    = bh >> 4;
    const int h    = bh & 15;
    const int q0   = (blockIdx.x * 4 + wave) * 16;

    const int nlo  = lane & 15;
    const int half = lane >> 4;

    const float*  P     = attn + (size_t)bh * SEQ * SEQ;
    const float*  parow = P + (size_t)(q0 + nlo) * SEQ + (half << 4);
    const __bf16* vbase = Vt + (size_t)(h * DK + nlo) * MTOT + (size_t)b * SEQ + (half << 4);

    f32x8 acc0 = {}, acc1 = {}, acc2 = {}, acc3 = {};
#pragma unroll 2
    for (int kk0 = 0; kk0 < SEQ; kk0 += 32) {
        const float* pa = parow + kk0;
        f32x4 a0 = *(const f32x4*)(pa);
        f32x4 a1 = *(const f32x4*)(pa + 4);
        f32x4 a2 = *(const f32x4*)(pa + 8);
        f32x4 a3 = *(const f32x4*)(pa + 12);
        const __bf16* pv = vbase + kk0;
        bf16x16 b0 = *(const bf16x16*)(pv);
        bf16x16 b1 = *(const bf16x16*)(pv + 16 * MTOT);
        bf16x16 b2 = *(const bf16x16*)(pv + 32 * MTOT);
        bf16x16 b3 = *(const bf16x16*)(pv + 48 * MTOT);
        bf16x16 a;
#pragma unroll
        for (int i = 0; i < 4; ++i) {
            a[i]      = (__bf16)a0[i];
            a[4 + i]  = (__bf16)a1[i];
            a[8 + i]  = (__bf16)a2[i];
            a[12 + i] = (__bf16)a3[i];
        }
        acc0 = WMMA_BF16(acc0, a, b0);
        acc1 = WMMA_BF16(acc1, a, b1);
        acc2 = WMMA_BF16(acc2, a, b2);
        acc3 = WMMA_BF16(acc3, a, b3);
    }

    const int mb = q0 + (half << 3);
    f32x8 accs[4] = {acc0, acc1, acc2, acc3};
#pragma unroll
    for (int t = 0; t < 4; ++t) {
        const int d = h * DK + 16 * t + nlo;
#pragma unroll
        for (int r = 0; r < 8; ++r)
            Cx[(size_t)(b * SEQ + mb + r) * D_MODEL + d] = (__bf16)accs[t][r];
    }
}

// ---------------------------------------------------------------------------
// Kernel 5: out = ctx @ w_o.T + b_o  (f32 into d_out[0 .. 4M))
// ---------------------------------------------------------------------------
__global__ __launch_bounds__(128) void k_oproj(
    const __bf16* __restrict__ Cx, const __bf16* __restrict__ wob,
    const float* __restrict__ bo, float* __restrict__ out)
{
    const int lane = threadIdx.x & 31;
    const int wave = threadIdx.x >> 5;
    const int m0   = blockIdx.x * 16;
    const int n0   = blockIdx.y * 256 + wave * 64;

    const int nlo  = lane & 15;
    const int half = lane >> 4;

    f32x8 acc0 = {}, acc1 = {}, acc2 = {}, acc3 = {};
    const __bf16* wrow = wob + (size_t)(n0 + nlo) * D_MODEL + (half << 4);
    for (int k0 = 0; k0 < D_MODEL; k0 += 32) {
        bf16x16 a  = load_A_bf16(Cx, D_MODEL, m0, k0, lane);
        bf16x16 b0 = *(const bf16x16*)(wrow + k0);
        bf16x16 b1 = *(const bf16x16*)(wrow + k0 + 16 * D_MODEL);
        bf16x16 b2 = *(const bf16x16*)(wrow + k0 + 32 * D_MODEL);
        bf16x16 b3 = *(const bf16x16*)(wrow + k0 + 48 * D_MODEL);
        acc0 = WMMA_BF16(acc0, a, b0);
        acc1 = WMMA_BF16(acc1, a, b1);
        acc2 = WMMA_BF16(acc2, a, b2);
        acc3 = WMMA_BF16(acc3, a, b3);
    }

    const int mb = m0 + (half << 3);
    f32x8 accs[4] = {acc0, acc1, acc2, acc3};
#pragma unroll
    for (int t = 0; t < 4; ++t) {
        const int   n  = n0 + 16 * t + nlo;
        const float bb = bo[n];
#pragma unroll
        for (int r = 0; r < 8; ++r)
            out[(size_t)(mb + r) * D_MODEL + n] = accs[t][r] + bb;
    }
}

// ---------------------------------------------------------------------------
extern "C" void kernel_launch(void* const* d_in, const int* in_sizes, int n_in,
                              void* d_out, int out_size, void* d_ws, size_t ws_size,
                              hipStream_t stream)
{
    (void)in_sizes; (void)n_in; (void)out_size; (void)ws_size;

    const float* x  = (const float*)d_in[0];
    const float* wq = (const float*)d_in[1];
    const float* bq = (const float*)d_in[2];
    const float* wk = (const float*)d_in[3];
    const float* bk = (const float*)d_in[4];
    const float* wv = (const float*)d_in[5];
    const float* bv = (const float*)d_in[6];
    const float* wo = (const float*)d_in[7];
    const float* bo = (const float*)d_in[8];

    float* out  = (float*)d_out;
    float* attn = out + (size_t)MTOT * D_MODEL;

    char* ws = (char*)d_ws;
    const size_t MB = 1024 * 1024;
    __bf16* Qb  = (__bf16*)(ws);              //  8 MB
    __bf16* Kb  = (__bf16*)(ws + 8  * MB);    //  8 MB
    __bf16* Vt  = (__bf16*)(ws + 16 * MB);    //  8 MB (transposed V)
    __bf16* Cx  = (__bf16*)(ws + 24 * MB);    //  8 MB
    __bf16* xb  = (__bf16*)(ws + 32 * MB);    //  8 MB
    __bf16* wqb = (__bf16*)(ws + 40 * MB);    //  2 MB
    __bf16* wkb = (__bf16*)(ws + 42 * MB);    //  2 MB
    __bf16* wvb = (__bf16*)(ws + 44 * MB);    //  2 MB
    __bf16* wob = (__bf16*)(ws + 46 * MB);    //  2 MB

    // 0) one-time f32 -> bf16 conversions
    const int nX4 = (MTOT * D_MODEL) / 4;       // 1,048,576
    const int nW4 = (D_MODEL * D_MODEL) / 4;    //   262,144
    k_cvt_bf16<<<dim3((nX4 + 255) / 256), 256, 0, stream>>>(x,  xb,  nX4);
    k_cvt_bf16<<<dim3((nW4 + 255) / 256), 256, 0, stream>>>(wq, wqb, nW4);
    k_cvt_bf16<<<dim3((nW4 + 255) / 256), 256, 0, stream>>>(wk, wkb, nW4);
    k_cvt_bf16<<<dim3((nW4 + 255) / 256), 256, 0, stream>>>(wv, wvb, nW4);
    k_cvt_bf16<<<dim3((nW4 + 255) / 256), 256, 0, stream>>>(wo, wob, nW4);

    // 1) QKV projections
    k_qkv_proj<<<dim3(MTOT / 16, D_MODEL / 256, 3), 128, 0, stream>>>(
        xb, wqb, bq, wkb, bk, wvb, bv, Qb, Kb, Vt);

    // 2) raw scaled scores (TDM-staged K tiles)
    k_scores<<<dim3(SEQ / 16, SEQ / 256, NBATCH * NHEADS), 128, 0, stream>>>(
        Qb, Kb, attn);

    // 3) softmax in place (65536 rows)
    k_softmax<<<dim3(NBATCH * NHEADS * SEQ), 256, 0, stream>>>(attn);

    // 4) ctx = attn @ V
    k_pv<<<dim3(SEQ / 64, NBATCH * NHEADS), 128, 0, stream>>>(attn, Vt, Cx);

    // 5) output projection
    k_oproj<<<dim3(MTOT / 16, D_MODEL / 256, 1), 128, 0, stream>>>(Cx, wob, bo, out);
}
